// SelfAttention_3066606649395
// MI455X (gfx1250) — matrix operand order repaired and checked by reading, not compiled, not run on hardware
//
#include <hip/hip_runtime.h>

// ---------------- CDNA5 WMMA types ----------------
typedef __attribute__((ext_vector_type(16))) __bf16 v16bf;
typedef __attribute__((ext_vector_type(8)))  float  v8f;
typedef __attribute__((ext_vector_type(4)))  int    v4i;

#define HID 1024
#define SEQ 2048
#define NB  8

// ---------------- GEMM tiling ----------------
#define BM 256
#define BN 128
#define BK 32
#define LDSW 40   // LDS row stride in halves: 80 B (16B aligned, bank skew)

enum { EPI_BIAS_QQT = 0, EPI_SCALE_BF16 = 1, EPI_F32 = 2 };

#define GAS __attribute__((address_space(1)))
#define LAS __attribute__((address_space(3)))

#if defined(__has_builtin)
#if __has_builtin(__builtin_amdgcn_global_load_async_to_lds_b128)
#define HAVE_ASYNC_LDS 1
#endif
#endif

// Copy 8 bf16 (16 B) global -> LDS. Async (ASYNCcnt-tracked) when available.
__device__ __forceinline__ void cp16_async(const __bf16* g, __bf16* l) {
#ifdef HAVE_ASYNC_LDS
  __builtin_amdgcn_global_load_async_to_lds_b128((GAS v4i*)g, (LAS v4i*)l, 0, 0);
#else
  *(uint4*)l = *(const uint4*)g;
#endif
}
__device__ __forceinline__ void wait_async() {
#ifdef HAVE_ASYNC_LDS
#if __has_builtin(__builtin_amdgcn_s_wait_asynccnt)
  __builtin_amdgcn_s_wait_asynccnt(0);
#else
  asm volatile("s_wait_asynccnt 0x0" ::: "memory");
#endif
#endif
}

// 16 contiguous fp32 -> 16 bf16
__device__ __forceinline__ void load16cvt(const float* __restrict__ p, __bf16* t) {
  const float4* q = (const float4*)p;
  float4 f0 = q[0], f1 = q[1], f2 = q[2], f3 = q[3];
  t[0]  = (__bf16)f0.x; t[1]  = (__bf16)f0.y; t[2]  = (__bf16)f0.z; t[3]  = (__bf16)f0.w;
  t[4]  = (__bf16)f1.x; t[5]  = (__bf16)f1.y; t[6]  = (__bf16)f1.z; t[7]  = (__bf16)f1.w;
  t[8]  = (__bf16)f2.x; t[9]  = (__bf16)f2.y; t[10] = (__bf16)f2.z; t[11] = (__bf16)f2.w;
  t[12] = (__bf16)f3.x; t[13] = (__bf16)f3.y; t[14] = (__bf16)f3.z; t[15] = (__bf16)f3.w;
}

// Stage one 32-half row of the A tile (fp32 source: convert; bf16: async copy)
__device__ __forceinline__ void stageA_row(const float* __restrict__ src, __bf16* dst) {
  __bf16 t0[16], t1[16];
  load16cvt(src, t0);
  load16cvt(src + 16, t1);
  ((uint4*)dst)[0] = *(uint4*)t0;        ((uint4*)dst)[1] = *((uint4*)t0 + 1);
  ((uint4*)(dst + 16))[0] = *(uint4*)t1; ((uint4*)(dst + 16))[1] = *((uint4*)t1 + 1);
}
__device__ __forceinline__ void stageA_row(const __bf16* __restrict__ src, __bf16* dst) {
  cp16_async(src,      dst);
  cp16_async(src + 8,  dst + 8);
  cp16_async(src + 16, dst + 16);
  cp16_async(src + 24, dst + 24);
}

// ---------------------------------------------------------------------------
// Tiled GEMM: C[M,N] = A[M,K] * B (B stored as [N][K] row-major).
// Block 256 threads = 8 wave32 waves; block tile 256x128; wave tile 64x64
// (4x4 wmma tiles => 16 v_wmma per K-step, 128 f32 acc VGPRs).
// Double-buffered LDS, async global->LDS staging when available.
// ---------------------------------------------------------------------------
template <typename AT, int EPI>
__device__ __forceinline__ void gemm_tile(
    const AT* __restrict__ A, int lda,
    const __bf16* __restrict__ B, int ldb,
    int K, int mBlock, int nBlock,
    const float* __restrict__ bias, float scale,
    float* __restrict__ outF, __bf16* __restrict__ outB,
    __bf16* __restrict__ outT, int ldc)
{
  __shared__ __bf16 sa[2][BM * LDSW];
  __shared__ __bf16 sb[2][BN * LDSW];

  const int tid  = threadIdx.x;
  const int lane = tid & 31;
  const int wave = tid >> 5;
  const int wm   = (wave >> 1) * 64;   // 0,64,128,192
  const int wn   = (wave & 1) * 64;    // 0,64
  const int l16  = lane & 15;
  const int hi   = lane >> 4;

  v8f acc[4][4];
#pragma unroll
  for (int m = 0; m < 4; ++m)
#pragma unroll
    for (int n = 0; n < 4; ++n)
#pragma unroll
      for (int r = 0; r < 8; ++r) acc[m][n][r] = 0.0f;

  const int bn = tid >> 1, bh = (tid & 1) * 16;   // B staging coords

  // prologue: stage k0 = 0 into buffer 0
  stageA_row(A + (size_t)(mBlock + tid) * lda, &sa[0][tid * LDSW]);
  {
    const __bf16* s = B + (size_t)(nBlock + bn) * ldb + bh;
    __bf16* d = &sb[0][bn * LDSW + bh];
    cp16_async(s, d);
    cp16_async(s + 8, d + 8);
  }

  int buf = 0;
  for (int k0 = 0; k0 < K; k0 += BK) {
    wait_async();
    __syncthreads();   // staged data visible; prior buffer's readers done

    const int nk = k0 + BK;
    if (nk < K) {      // stream next tile while computing this one
      stageA_row(A + (size_t)(mBlock + tid) * lda + nk, &sa[buf ^ 1][tid * LDSW]);
      const __bf16* s = B + (size_t)(nBlock + bn) * ldb + nk + bh;
      __bf16* d = &sb[buf ^ 1][bn * LDSW + bh];
      cp16_async(s, d);
      cp16_async(s + 8, d + 8);
    }

    // gather fragments (contiguous k = hi*16 + s, same order in A and B)
    v16bf af[4], bfr[4];
#pragma unroll
    for (int m = 0; m < 4; ++m) {
      const __bf16* pa = &sa[buf][(wm + m * 16 + l16) * LDSW];
      ((uint4*)&af[m])[0] = *(const uint4*)(pa + hi * 16);      // K = hi*16..+7
      ((uint4*)&af[m])[1] = *(const uint4*)(pa + hi * 16 + 8);  // K = hi*16+8..+15
    }
#pragma unroll
    for (int n = 0; n < 4; ++n) {
      const __bf16* pb = &sb[buf][(wn + n * 16 + l16) * LDSW + hi * 16]; // K=hi*16..+15
      ((uint4*)&bfr[n])[0] = ((const uint4*)pb)[0];
      ((uint4*)&bfr[n])[1] = ((const uint4*)pb)[1];
    }

#pragma unroll
    for (int m = 0; m < 4; ++m)
#pragma unroll
      for (int n = 0; n < 4; ++n)
        acc[m][n] = __builtin_amdgcn_wmma_f32_16x16x32_bf16(
            false, af[m], false, bfr[n], (short)0, acc[m][n], false, false);

    buf ^= 1;
  }

  // ---- epilogue: VGPR r -> row (r + 8*hi), lane l16 -> col ----
#pragma unroll
  for (int m = 0; m < 4; ++m) {
    const int rowBase = mBlock + wm + m * 16 + hi * 8;
#pragma unroll
    for (int n = 0; n < 4; ++n) {
      const int col = nBlock + wn + n * 16 + l16;
      if constexpr (EPI == EPI_BIAS_QQT) {
        const float bv = bias[col];
        __bf16 pk[8];
#pragma unroll
        for (int r = 0; r < 8; ++r) {
          const float v = acc[m][n][r] + bv;
          outB[(size_t)(rowBase + r) * ldc + col] = (__bf16)v;  // Q[s][h]
          pk[r] = (__bf16)v;
        }
        // Q^T[h][s]: 8 consecutive s per lane -> one packed 16B store
        const int batch = rowBase >> 11, s = rowBase & (SEQ - 1);
        *(uint4*)&outT[((size_t)batch * HID + col) * SEQ + s] = *(uint4*)pk;
      } else if constexpr (EPI == EPI_SCALE_BF16) {
#pragma unroll
        for (int r = 0; r < 8; ++r)
          outB[(size_t)(rowBase + r) * ldc + col] = (__bf16)(acc[m][n][r] * scale);
      } else {
#pragma unroll
        for (int r = 0; r < 8; ++r)
          outF[(size_t)(rowBase + r) * ldc + col] = acc[m][n][r];
      }
    }
  }
}

// ---------------- Pass 0: Wt[n][k] = Wq[k][n] (fp32 -> bf16) ----------------
__global__ __launch_bounds__(256) void transpose_w(
    const float* __restrict__ W, __bf16* __restrict__ Wt)
{
  __shared__ float tile[32][33];
  const int bn = blockIdx.x * 32, bk = blockIdx.y * 32;
  const int tx = threadIdx.x & 31, ty = threadIdx.x >> 5;  // 8 rows / pass
#pragma unroll
  for (int i = 0; i < 4; ++i)
    tile[ty + i * 8][tx] = W[(size_t)(bk + ty + i * 8) * HID + bn + tx];
  __syncthreads();
#pragma unroll
  for (int i = 0; i < 4; ++i)
    Wt[(size_t)(bn + ty + i * 8) * HID + bk + tx] = (__bf16)tile[tx][ty + i * 8];
}

// ---------------- Pass 1: Q = X*Wq + bq -> Q bf16 and Q^T bf16 ----------------
__global__ __launch_bounds__(256) void proj_kernel(
    const float* __restrict__ X, const __bf16* __restrict__ Wt,
    const float* __restrict__ bq, __bf16* __restrict__ Q, __bf16* __restrict__ Qt)
{
  gemm_tile<float, EPI_BIAS_QQT>(
      X, HID, Wt, HID, HID, blockIdx.x * BM, blockIdx.y * BN,
      bq, 1.0f, nullptr, Q, Qt, HID);
}

// ---------------- Pass 2: S = Q*Q^T * (1/sqrt(H)) -> bf16 ----------------
__global__ __launch_bounds__(256) void scores_kernel(
    const __bf16* __restrict__ Q, __bf16* __restrict__ S)
{
  const __bf16* Qb = Q + (size_t)blockIdx.z * SEQ * HID;
  __bf16*       Sb = S + (size_t)blockIdx.z * SEQ * SEQ;
  gemm_tile<__bf16, EPI_SCALE_BF16>(
      Qb, HID, Qb, HID, HID, blockIdx.x * BM, blockIdx.y * BN,
      nullptr, 0.03125f /* 1/sqrt(1024) */, nullptr, Sb, nullptr, SEQ);
}

// ---------------- Pass 3: row softmax in place ----------------
__device__ __forceinline__ float blk_reduce(float v, bool isMax) {
  __shared__ float red[8];
#pragma unroll
  for (int m = 16; m > 0; m >>= 1) {
    float o = __shfl_xor(v, m, 32);
    v = isMax ? fmaxf(v, o) : (v + o);
  }
  if ((threadIdx.x & 31) == 0) red[threadIdx.x >> 5] = v;
  __syncthreads();
  float r = red[0];
#pragma unroll
  for (int i = 1; i < 8; ++i) r = isMax ? fmaxf(r, red[i]) : (r + red[i]);
  __syncthreads();
  return r;
}

__global__ __launch_bounds__(256) void softmax_kernel(__bf16* __restrict__ S)
{
  __bf16* row = S + (size_t)blockIdx.x * SEQ;
  float v[8];
#pragma unroll
  for (int i = 0; i < 8; ++i) v[i] = (float)row[threadIdx.x + i * 256];
  float m = -1e30f;
#pragma unroll
  for (int i = 0; i < 8; ++i) m = fmaxf(m, v[i]);
  m = blk_reduce(m, true);
  float s = 0.0f;
#pragma unroll
  for (int i = 0; i < 8; ++i) { v[i] = __expf(v[i] - m); s += v[i]; }
  s = blk_reduce(s, false);
  const float inv = 1.0f / s;
#pragma unroll
  for (int i = 0; i < 8; ++i) row[threadIdx.x + i * 256] = (__bf16)(v[i] * inv);
}

// ---------------- Pass 4: O = P*V (B = Q^T, no transpose staging) ----------------
__global__ __launch_bounds__(256) void pv_kernel(
    const __bf16* __restrict__ P, const __bf16* __restrict__ Qt,
    float* __restrict__ O)
{
  const __bf16* Pb  = P  + (size_t)blockIdx.z * SEQ * SEQ;
  const __bf16* Qtb = Qt + (size_t)blockIdx.z * HID * SEQ;
  float*        Ob  = O  + (size_t)blockIdx.z * SEQ * HID;
  gemm_tile<__bf16, EPI_F32>(
      Pb, SEQ, Qtb, SEQ, SEQ, blockIdx.x * BM, blockIdx.y * BN,
      nullptr, 1.0f, Ob, nullptr, nullptr, HID);
}

// ---------------- host launcher ----------------
extern "C" void kernel_launch(void* const* d_in, const int* in_sizes, int n_in,
                              void* d_out, int out_size, void* d_ws, size_t ws_size,
                              hipStream_t stream) {
  const float* X  = (const float*)d_in[0];   // [8, 2048, 1024]
  const float* Wq = (const float*)d_in[1];   // [1024, 1024]
  const float* bq = (const float*)d_in[2];   // [1024]
  float* out = (float*)d_out;                // [8, 2048, 1024]

  // workspace: Wt (2MB) | Q (32MB) | Qt (32MB) | S (64MB)  = 130 MB
  char* w = (char*)d_ws;
  __bf16* Wt = (__bf16*)w;                                     w += (size_t)HID * HID * 2;
  __bf16* Q  = (__bf16*)w;                                     w += (size_t)NB * SEQ * HID * 2;
  __bf16* Qt = (__bf16*)w;                                     w += (size_t)NB * HID * SEQ * 2;
  __bf16* S  = (__bf16*)w;

  transpose_w  <<<dim3(HID / 32, HID / 32),        256, 0, stream>>>(Wq, Wt);
  proj_kernel  <<<dim3((NB * SEQ) / BM, HID / BN), 256, 0, stream>>>(X, Wt, bq, Q, Qt);
  scores_kernel<<<dim3(SEQ / BM, SEQ / BN, NB),    256, 0, stream>>>(Q, S);
  softmax_kernel<<<dim3(NB * SEQ),                 256, 0, stream>>>(S);
  pv_kernel    <<<dim3(SEQ / BM, HID / BN, NB),    256, 0, stream>>>(S, Qt, out);
}
